// CustomLinear_9620726743744
// MI455X (gfx1250) — matrix operand order repaired and checked by reading, not compiled, not run on hardware
//
#include <hip/hip_runtime.h>
#include <hip/hip_bf16.h>

// M=8192 (4*2048), K=4096 (IN_FEATURES), N=4096 (OUT_FEATURES)
// Workspace (192 MiB):
//   Wt_hi [N][K] bf16 (32MB) | Wt_lo [N][K] bf16 (32MB)
//   X_hi  [M][K] bf16 (64MB) | X_lo  [M][K] bf16 (64MB)

#define KDIM 4096
#define NDIM 4096
#define MDIM 8192

typedef __attribute__((ext_vector_type(16))) __bf16 bf16x16;
typedef __attribute__((ext_vector_type(8)))  __bf16 bf16x8;
typedef __attribute__((ext_vector_type(8)))  float  floatx8;

// ---------------------------------------------------------------------------
// Prologue 1: gather cluster vectors -> W[k][n], split fp32 into bf16 hi/lo,
// store TRANSPOSED (Wt[n][k]) so GEMM B-fragments are contiguous per lane.
// LDS-tiled (64k x 64n) so index reads and global stores are coalesced.
// ---------------------------------------------------------------------------
__global__ __launch_bounds__(256) void gather_split_kernel(
    const float* __restrict__ cluster, const int* __restrict__ index,
    __bf16* __restrict__ wt_hi, __bf16* __restrict__ wt_lo) {
  __shared__ __align__(16) __bf16 sh[2][64][72];  // [hi/lo][n_local][k_local], +8 pad
  const int nBase = blockIdx.x * 64;
  const int kBase = blockIdx.y * 64;
  const int t = threadIdx.x;

#pragma unroll
  for (int e0 = 0; e0 < 1024; e0 += 256) {
    const int e  = e0 + t;
    const int kk = e >> 4;        // 0..63  (k_local)
    const int g  = e & 15;        // 0..15  (n group of 4)
    const int k  = kBase + kk;
    const int idx = index[k * (NDIM / 4) + (nBase >> 2) + g];
    const float4 cv = *(const float4*)(cluster + (size_t)idx * 4);
    const float vals[4] = {cv.x, cv.y, cv.z, cv.w};
#pragma unroll
    for (int j = 0; j < 4; ++j) {
      const float v = vals[j];
      const __bf16 h = (__bf16)v;
      sh[0][g * 4 + j][kk] = h;
      sh[1][g * 4 + j][kk] = (__bf16)(v - (float)h);
    }
  }
  __syncthreads();

#pragma unroll
  for (int c0 = 0; c0 < 1024; c0 += 256) {
    const int c   = c0 + t;
    const int mat = c >> 9;        // 0 = hi, 1 = lo
    const int r   = (c >> 3) & 63; // n_local row
    const int ch  = c & 7;         // 16B chunk within 64-k (128B) row
    const uint4 dat = *(const uint4*)&sh[mat][r][ch * 8];
    __bf16* dst = (mat ? wt_lo : wt_hi) +
                  (size_t)(nBase + r) * KDIM + kBase + ch * 8;
    *(uint4*)dst = dat;
  }
}

// ---------------------------------------------------------------------------
// Prologue 2: split x (fp32, row-major M x K) into bf16 hi/lo planes.
// One streaming pass: 8 elements per thread, 16B stores.
// ---------------------------------------------------------------------------
__global__ __launch_bounds__(256) void split_x_kernel(
    const float* __restrict__ x, __bf16* __restrict__ xh,
    __bf16* __restrict__ xl) {
  const size_t base = ((size_t)blockIdx.x * 256 + threadIdx.x) * 8;
  const float4 a = *(const float4*)(x + base);
  const float4 b = *(const float4*)(x + base + 4);
  const float f[8] = {a.x, a.y, a.z, a.w, b.x, b.y, b.z, b.w};
  bf16x8 h, l;
#pragma unroll
  for (int i = 0; i < 8; ++i) {
    const __bf16 hv = (__bf16)f[i];
    h[i] = hv;
    l[i] = (__bf16)(f[i] - (float)hv);
  }
  *(bf16x8*)(xh + base) = h;
  *(bf16x8*)(xl + base) = l;
}

// ---------------------------------------------------------------------------
// GEMM: block = 256 threads = 8 waves (4m x 2n), block tile 128(M) x 128(N),
// wave tile 32x64 = 2x4 WMMA accumulators. 3-product bf16 split per tile:
// 24 v_wmma per K-step vs 24 b128 loads, zero conversion VALU in the loop.
// ---------------------------------------------------------------------------
__device__ __forceinline__ bf16x16 load_a_frag(const __bf16* __restrict__ p) {
  // A k order used here: elements 0..15 = K off..off+15 (contiguous)
  const bf16x8 c0 = *(const bf16x8*)(p);
  const bf16x8 c1 = *(const bf16x8*)(p + 8);
  return __builtin_shufflevector(c0, c1, 0, 1, 2, 3, 4, 5, 6, 7,
                                 8, 9, 10, 11, 12, 13, 14, 15);
}

__global__ __launch_bounds__(256) void gemm_bf16split_kernel(
    const __bf16* __restrict__ xh, const __bf16* __restrict__ xl,
    const __bf16* __restrict__ wt_hi, const __bf16* __restrict__ wt_lo,
    const float* __restrict__ bias, float* __restrict__ out) {
  const int lane   = threadIdx.x & 31;
  const int wave   = threadIdx.x >> 5;
  const int wave_m = wave >> 1;  // 0..3
  const int wave_n = wave & 1;   // 0..1
  const int mBase  = blockIdx.y * 128 + wave_m * 32;
  const int nBase  = blockIdx.x * 128 + wave_n * 64;
  const int l15    = lane & 15;
  const int aoff   = (lane < 16) ? 0 : 16;  // A K-offset: contiguous k = 16h + s
  const int boff   = (lane < 16) ? 0 : 16;  // B K-offset: contiguous k = 16h + s (same as A)

  const __bf16* ahp[2];
  const __bf16* alp[2];
#pragma unroll
  for (int mt = 0; mt < 2; ++mt) {
    const size_t ro = (size_t)(mBase + mt * 16 + l15) * KDIM + aoff;
    ahp[mt] = xh + ro;
    alp[mt] = xl + ro;
  }
  const __bf16* bhp[4];
  const __bf16* blp[4];
#pragma unroll
  for (int nt = 0; nt < 4; ++nt) {
    const size_t ro = (size_t)(nBase + nt * 16 + l15) * KDIM + boff;
    bhp[nt] = wt_hi + ro;
    blp[nt] = wt_lo + ro;
  }

  const floatx8 zero = {0.f, 0.f, 0.f, 0.f, 0.f, 0.f, 0.f, 0.f};
  floatx8 acc[2][4];
#pragma unroll
  for (int mt = 0; mt < 2; ++mt)
#pragma unroll
    for (int nt = 0; nt < 4; ++nt) acc[mt][nt] = zero;

  for (int k0 = 0; k0 < KDIM; k0 += 32) {
    if (k0 + 128 < KDIM) {  // pull next tiles toward the WGP (global_prefetch_b8)
      __builtin_prefetch(ahp[0] + k0 + 128, 0, 0);
      __builtin_prefetch(bhp[0] + k0 + 128, 0, 0);
    }
    bf16x16 ahi[2], alo[2];
#pragma unroll
    for (int mt = 0; mt < 2; ++mt) {
      ahi[mt] = load_a_frag(ahp[mt] + k0);
      alo[mt] = load_a_frag(alp[mt] + k0);
    }
    bf16x16 bhi[4], blo[4];
#pragma unroll
    for (int nt = 0; nt < 4; ++nt) {
      bhi[nt] = *(const bf16x16*)(bhp[nt] + k0);
      blo[nt] = *(const bf16x16*)(blp[nt] + k0);
    }
#pragma unroll
    for (int mt = 0; mt < 2; ++mt) {
#pragma unroll
      for (int nt = 0; nt < 4; ++nt) {
        acc[mt][nt] = __builtin_amdgcn_wmma_f32_16x16x32_bf16(
            false, ahi[mt], false, bhi[nt], (short)0, acc[mt][nt], false, false);
        acc[mt][nt] = __builtin_amdgcn_wmma_f32_16x16x32_bf16(
            false, ahi[mt], false, blo[nt], (short)0, acc[mt][nt], false, false);
        acc[mt][nt] = __builtin_amdgcn_wmma_f32_16x16x32_bf16(
            false, alo[mt], false, bhi[nt], (short)0, acc[mt][nt], false, false);
      }
    }
  }

  // Epilogue: C layout -> VGPR r holds rows r (lanes 0-15) / r+8 (lanes 16-31),
  // col = lane%16. Add bias, store fp32.
  const int rowAdd = (lane < 16) ? 0 : 8;
#pragma unroll
  for (int nt = 0; nt < 4; ++nt) {
    const int col = nBase + nt * 16 + l15;
    const float bv = bias[col];
#pragma unroll
    for (int mt = 0; mt < 2; ++mt) {
#pragma unroll
      for (int r = 0; r < 8; ++r) {
        const int row = mBase + mt * 16 + r + rowAdd;
        out[(size_t)row * NDIM + col] = acc[mt][nt][r] + bv;
      }
    }
  }
}

// ---------------------------------------------------------------------------
extern "C" void kernel_launch(void* const* d_in, const int* in_sizes, int n_in,
                              void* d_out, int out_size, void* d_ws, size_t ws_size,
                              hipStream_t stream) {
  const float* x       = (const float*)d_in[0];   // (4,2048,4096) f32
  const float* cluster = (const float*)d_in[1];   // (65536,4) f32
  const int*   index   = (const int*)d_in[2];     // (4194304,) i32
  const float* bias    = (const float*)d_in[3];   // (4096,) f32
  float* out = (float*)d_out;                      // (4,2048,4096) f32

  __bf16* wt_hi = (__bf16*)d_ws;                           // 32 MB
  __bf16* wt_lo = wt_hi + (size_t)NDIM * KDIM;             // 32 MB
  __bf16* x_hi  = wt_lo + (size_t)NDIM * KDIM;             // 64 MB
  __bf16* x_lo  = x_hi + (size_t)MDIM * KDIM;              // 64 MB

  // 1) Gather + fp32 -> bf16 hi/lo split, transposed weight (N x K).
  gather_split_kernel<<<dim3(NDIM / 64, KDIM / 64), 256, 0, stream>>>(
      cluster, index, wt_hi, wt_lo);

  // 2) Split x into bf16 hi/lo planes (streaming pass, ~11us at HBM speed).
  split_x_kernel<<<(MDIM * (size_t)KDIM) / (256 * 8), 256, 0, stream>>>(
      x, x_hi, x_lo);

  // 3) Split-bf16 WMMA GEMM: grid = (N/128, M/128), 24 wmma / K-step / wave.
  gemm_bf16split_kernel<<<dim3(NDIM / 128, MDIM / 128), 256, 0, stream>>>(
      x_hi, x_lo, wt_hi, wt_lo, bias, out);
}
